// Recurrent_Attn_2095944041172
// MI455X (gfx1250) — matrix operand order repaired and checked by reading, not compiled, not run on hardware
//
#include <hip/hip_runtime.h>
#include <hip/hip_bf16.h>
#include <cstddef>
#include <cstdint>

typedef __bf16 bf16;
typedef __attribute__((ext_vector_type(16))) __bf16 v16bf;
typedef __attribute__((ext_vector_type(8)))  __bf16 bf16x8;
typedef __attribute__((ext_vector_type(8)))  float   v8f;
typedef __attribute__((ext_vector_type(4)))  float   f32x4;

typedef __attribute__((ext_vector_type(4))) unsigned int u32x4;
typedef __attribute__((ext_vector_type(4))) int          i32x4;
typedef __attribute__((ext_vector_type(8))) int          i32x8;

#define CCH   256
#define HH    64
#define WW    64
#define HWS   4096      // 64*64
#define KTOT  2304      // 9*256
#define BM    64
#define BN    128
#define BK    64        // two 16x16x32 K-steps per buffer
#define NCHUNK 36       // KTOT/BK
#define WELEM 589824    // 256*2304 per weight tensor

#if __has_builtin(__builtin_amdgcn_tensor_load_to_lds)
#define USE_TDM 1
#else
#define USE_TDM 0
#endif

// ---------------------------------------------------------------------------
// Weight repack: OIHW fp32 -> wT[co][t*256+cin] bf16 (K contiguous per row)
// ---------------------------------------------------------------------------
__global__ __launch_bounds__(256) void wcvt_kernel(const float* __restrict__ w,
                                                   bf16* __restrict__ wT) {
    int idx = blockIdx.x * 256 + threadIdx.x;          // over 256*256*9
    int t   = idx % 9;
    int r   = idx / 9;
    int cin = r & (CCH - 1);
    int co  = r >> 8;
    wT[(size_t)co * KTOT + t * CCH + cin] = (bf16)w[idx];
}

// ---------------------------------------------------------------------------
// Activation convert: NCHW fp32 -> NHWC bf16
// ---------------------------------------------------------------------------
__global__ __launch_bounds__(256) void nchw2nhwc_kernel(const float* __restrict__ in,
                                                        bf16* __restrict__ out) {
    int idx = blockIdx.x * 256 + threadIdx.x;          // NCHW linear index
    int p  = idx & (HWS - 1);
    int bc = idx >> 12;
    int c  = bc & (CCH - 1);
    int b  = bc >> 8;
    out[((size_t)b * HWS + p) * CCH + c] = (bf16)in[idx];
}

// ---------------------------------------------------------------------------
// TDM: async-load a [128 rows x 64 bf16] tile of wT into LDS.
// D# per CDNA5 ISA ch.8: group0 = {flags, lds_addr, global_addr, type=2},
// group1 = {data_size=2B, tensor dims/strides, tile dims}.
// ---------------------------------------------------------------------------
#if USE_TDM
__device__ __forceinline__ void tdm_load_weights(unsigned lds_off,
                                                 const bf16* gsrc) {
    unsigned long long ga = (unsigned long long)(uintptr_t)gsrc;
    u32x4 g0;
    g0[0] = 1u;                                    // count=1 (valid, user mode)
    g0[1] = lds_off;                               // lds_addr (bytes)
    g0[2] = (unsigned)(ga & 0xffffffffu);          // global_addr[31:0]
    g0[3] = (unsigned)((ga >> 32) & 0x01ffffffu)   // global_addr[56:32]
          | (2u << 30);                            // type = 2 ("image")
    i32x8 g1;
    g1[0] = (int)(1u << 16);                       // data_size=1 (2 bytes)
    g1[1] = (int)((unsigned)(KTOT & 0xffff) << 16);// tensor_dim0[15:0] @bits63:48
    g1[2] = (int)((KTOT >> 16) & 0xffff)           // tensor_dim0[31:16]
          | (int)(256u << 16);                     // tensor_dim1[15:0]
    g1[3] = 0                                      // tensor_dim1[31:16]
          | (int)(64u << 16);                      // tile_dim0 = 64 (K slice)
    g1[4] = (int)128u;                             // tile_dim1 = 128 rows
    g1[5] = (int)KTOT;                             // tensor_dim0_stride[31:0]
    g1[6] = 0;                                     // stride hi / dim1_stride lo
    g1[7] = 0;
    i32x4 z4 = {0, 0, 0, 0};
#if defined(__clang_major__) && (__clang_major__ >= 23)
    i32x8 z8 = {0, 0, 0, 0, 0, 0, 0, 0};
    __builtin_amdgcn_tensor_load_to_lds(g0, g1, z4, z4, z8, 0);
#else
    __builtin_amdgcn_tensor_load_to_lds(g0, g1, z4, z4, 0);
#endif
}
__device__ __forceinline__ void tdm_wait0() {
#if __has_builtin(__builtin_amdgcn_s_wait_tensorcnt)
    __builtin_amdgcn_s_wait_tensorcnt(0);
#else
    asm volatile("s_wait_tensorcnt 0x0" ::: "memory");
#endif
}
#endif

// ---------------------------------------------------------------------------
// Implicit-GEMM conv3x3 (stride 1, pad=dil, dilation dil) + BN + LeakyReLU.
// M = pixels (NHWC bf16 input), N = out channels, K = tap*256+cin.
// ---------------------------------------------------------------------------
__global__ __launch_bounds__(256, 2)
void conv3x3_wmma_kernel(const bf16* __restrict__ inNHWC,
                         const bf16* __restrict__ w0, const bf16* __restrict__ w1,
                         const bf16* __restrict__ w2,
                         const float* __restrict__ s0, const float* __restrict__ s1,
                         const float* __restrict__ s2,
                         const float* __restrict__ bb0, const float* __restrict__ bb1,
                         const float* __restrict__ bb2,
                         float* __restrict__ o0, float* __restrict__ o1,
                         float* __restrict__ o2,
                         int dil) {
    __shared__ __align__(16) bf16 As[2][BM][BK];   // 16 KB
    __shared__ __align__(16) bf16 Bs[2][BN][BK];   // 32 KB

    const int z = blockIdx.z;
    const bf16*  wT    = (z == 0) ? w0  : ((z == 1) ? w1  : w2);
    const float* scale = (z == 0) ? s0  : ((z == 1) ? s1  : s2);
    const float* bias  = (z == 0) ? bb0 : ((z == 1) ? bb1 : bb2);
    float*       out   = (z == 0) ? o0  : ((z == 1) ? o1  : o2);

    const int tid  = threadIdx.x;
    const int lane = tid & 31;
    const int wid  = tid >> 5;
    const int wm   = wid & 1;        // 2 wave rows    -> 32 M each
    const int wn   = wid >> 1;       // 4 wave columns -> 32 N each

    const int mblk  = blockIdx.x;            // 0..255
    const int nbase = blockIdx.y * BN;       // 0 / 128
    const int b     = (mblk * BM) / HWS;     // tiles never straddle batches
    const int pbase = (mblk * BM) & (HWS - 1);

    // ---- A staging precompute (done once): pixel base + tap validity masks
    const int am   = tid >> 2;               // pixel row in tile (0..63)
    const int aseg = tid & 3;                // which 16-cin slice
    const int py   = (pbase + am) >> 6;
    const int px   = (pbase + am) & 63;
    const bf16* abase = inNHWC + ((size_t)b * HWS + py * WW + px) * CCH + aseg * 16;
    const int vy = ((py >= dil) ? 1 : 0) | 2 | ((py <= (HH - 1) - dil) ? 4 : 0);
    const int vx = ((px >= dil) ? 1 : 0) | 2 | ((px <= (WW - 1) - dil) ? 4 : 0);

    const unsigned bs_base = (unsigned)(uintptr_t)(const void*)&Bs[0][0][0];
    const bf16* wrow = wT + (size_t)nbase * KTOT;     // tile base in wT

    v8f c00 = {}, c01 = {}, c10 = {}, c11 = {};
    const int half = lane >> 4;
    const int l15  = lane & 15;

    auto stageA = [&](int buf, int kc) {
        int t    = kc >> 2;                  // 4 chunks of 64 per tap
        int cin0 = (kc & 3) * BK;
        int ty = (t * 11) >> 5;              // t/3 for t in 0..8
        int tx = t - 3 * ty;
        bf16x8 v0 = {}, v1 = {};
        if ((vy >> ty) & (vx >> tx) & 1) {
            const bf16* src = abase + dil * CCH * ((ty - 1) * WW + (tx - 1)) + cin0;
            v0 = *(const bf16x8*)src;
            v1 = *(const bf16x8*)(src + 8);
        }
        *(bf16x8*)&As[buf][am][aseg * 16]     = v0;
        *(bf16x8*)&As[buf][am][aseg * 16 + 8] = v1;
    };
#if !USE_TDM
    auto stageB = [&](int buf, int kc) {     // fallback: manual staging
        int row = tid >> 1, h32 = (tid & 1) * 32;
        const bf16* src = wrow + (size_t)row * KTOT + kc * BK + h32;
#pragma unroll
        for (int i = 0; i < 4; ++i)
            *(bf16x8*)&Bs[buf][row][h32 + i * 8] = *(const bf16x8*)(src + i * 8);
    };
#endif
    auto compute = [&](int buf) {
#pragma unroll
        for (int u = 0; u < 2; ++u) {        // two 16x16x32 K-steps
            union Frag { v16bf v; bf16x8 h[2]; };
            Frag a0, a1, fb0, fb1;
            const bf16* ar0 = &As[buf][wm * 32 + l15][u * 32];
            a0.h[0] = *(const bf16x8*)(ar0 + 8 * half);
            a0.h[1] = *(const bf16x8*)(ar0 + 16 + 8 * half);
            const bf16* ar1 = &As[buf][wm * 32 + 16 + l15][u * 32];
            a1.h[0] = *(const bf16x8*)(ar1 + 8 * half);
            a1.h[1] = *(const bf16x8*)(ar1 + 16 + 8 * half);
            const bf16* br0 = &Bs[buf][wn * 32 + l15][u * 32];
            fb0.h[0] = *(const bf16x8*)(br0 + 8 * half);
            fb0.h[1] = *(const bf16x8*)(br0 + 16 + 8 * half);
            const bf16* br1 = &Bs[buf][wn * 32 + 16 + l15][u * 32];
            fb1.h[0] = *(const bf16x8*)(br1 + 8 * half);
            fb1.h[1] = *(const bf16x8*)(br1 + 16 + 8 * half);
            c00 = __builtin_amdgcn_wmma_f32_16x16x32_bf16(false, a0.v, false, fb0.v,
                                                          (short)0, c00, false, false);
            c01 = __builtin_amdgcn_wmma_f32_16x16x32_bf16(false, a0.v, false, fb1.v,
                                                          (short)0, c01, false, false);
            c10 = __builtin_amdgcn_wmma_f32_16x16x32_bf16(false, a1.v, false, fb0.v,
                                                          (short)0, c10, false, false);
            c11 = __builtin_amdgcn_wmma_f32_16x16x32_bf16(false, a1.v, false, fb1.v,
                                                          (short)0, c11, false, false);
        }
    };

    // ---- prologue: fill buffer 0
    stageA(0, 0);
#if USE_TDM
    if (wid == 0) {
        tdm_load_weights(bs_base, wrow);
        tdm_wait0();
    }
#else
    stageB(0, 0);
#endif
    __syncthreads();

    // ---- main loop: stage (kc+1) async while computing kc
    for (int kc = 0; kc < NCHUNK; ++kc) {
        int cur = kc & 1;
        if (kc + 1 < NCHUNK) {
            stageA(cur ^ 1, kc + 1);
#if USE_TDM
            if (wid == 0)
                tdm_load_weights(bs_base + (unsigned)((cur ^ 1) * (BN * BK * 2)),
                                 wrow + (kc + 1) * BK);
#else
            stageB(cur ^ 1, kc + 1);
#endif
        }
        compute(cur);
#if USE_TDM
        if (wid == 0) tdm_wait0();           // next buffer's TDM complete
#endif
        __syncthreads();
    }

    // ---- epilogue: BN scale/bias + LeakyReLU(0.1), NCHW fp32 stores
    // C layout: VGPR r, lane -> M = r + 8*half, N = lane&15 => consecutive
    // r are consecutive pixels: two float4 stores per 16x16 tile.
#pragma unroll
    for (int mi = 0; mi < 2; ++mi) {
#pragma unroll
        for (int ni = 0; ni < 2; ++ni) {
            v8f acc = (mi == 0) ? ((ni == 0) ? c00 : c01)
                                : ((ni == 0) ? c10 : c11);
            int   co = nbase + wn * 32 + ni * 16 + l15;
            float sc = scale[co];
            float bi = bias[co];
            int   pm = pbase + wm * 32 + mi * 16 + 8 * half;
            float* op = out + ((size_t)b * CCH + co) * HWS + pm;
            f32x4 v0, v1;
#pragma unroll
            for (int r = 0; r < 8; ++r) {
                float v = acc[r] * sc + bi;
                v = (v > 0.0f) ? v : 0.1f * v;
                if (r < 4) v0[r] = v; else v1[r - 4] = v;
            }
            *(f32x4*)op       = v0;
            *(f32x4*)(op + 4) = v1;
        }
    }
}

// ---------------------------------------------------------------------------
// 3x3 neighborhood attention, per-channel softmax over 9 (zero-padded) taps.
// Matches reference: OOB taps contribute dot=0 (exp counted in denominator).
// ---------------------------------------------------------------------------
__global__ __launch_bounds__(256)
void attn_kernel(const float* __restrict__ Q, const float* __restrict__ K,
                 const float* __restrict__ V,
                 bf16* __restrict__ outNHWC, float* __restrict__ outNCHW) {
    int idx = blockIdx.x * 256 + threadIdx.x;   // NCHW linear
    int p  = idx & (HWS - 1);
    int bc = idx >> 12;
    int c  = bc & (CCH - 1);
    int b  = bc >> 8;
    int y  = p >> 6, x = p & 63;

    const float inv = 0.0625f;                  // 1/sqrt(256)
    float q = Q[idx];

    float d[9], vv[9];
#pragma unroll
    for (int j = 0; j < 9; ++j) {
        int dy = j / 3 - 1, dx = j % 3 - 1;
        int iy = y + dy, ix = x + dx;
        bool ok = ((unsigned)iy < (unsigned)HH) && ((unsigned)ix < (unsigned)WW);
        int  o  = idx + dy * WW + dx;
        float kk = ok ? K[o] : 0.0f;
        vv[j]    = ok ? V[o] : 0.0f;
        d[j]     = q * kk * inv;
    }
    float m = d[0];
#pragma unroll
    for (int j = 1; j < 9; ++j) m = fmaxf(m, d[j]);
    float den = 0.0f, num = 0.0f;
#pragma unroll
    for (int j = 0; j < 9; ++j) {
        float e = __expf(d[j] - m);
        den += e;
        num += e * vv[j];
    }
    float r = num / den;
    if (outNCHW) outNCHW[idx] = r;
    if (outNHWC) outNHWC[((size_t)b * HWS + p) * CCH + c] = (bf16)r;
}

// ---------------------------------------------------------------------------
// Host launcher
// ---------------------------------------------------------------------------
extern "C" void kernel_launch(void* const* d_in, const int* in_sizes, int n_in,
                              void* d_out, int out_size, void* d_ws, size_t ws_size,
                              hipStream_t stream) {
    (void)in_sizes; (void)n_in; (void)out_size; (void)ws_size;

    const float* x = (const float*)d_in[0];
    const float* w[6]; const float* sc[6]; const float* bi[6];
    for (int i = 0; i < 6; ++i) {               // q1,k1,v1,q2,k2,v2
        w[i]  = (const float*)d_in[1 + 3 * i];
        sc[i] = (const float*)d_in[2 + 3 * i];
        bi[i] = (const float*)d_in[3 + 3 * i];
    }

    char* ws = (char*)d_ws;
    bf16*  xh  = (bf16*)(ws);                       //  8 MB NHWC bf16
    bf16*  a1h = (bf16*)(ws + (8u  << 20));         //  8 MB NHWC bf16
    float* Qf  = (float*)(ws + (16u << 20));        // 16 MB NCHW f32
    float* Kf  = (float*)(ws + (32u << 20));        // 16 MB NCHW f32
    float* Vf  = (float*)(ws + (48u << 20));        // 16 MB NCHW f32
    bf16*  wT  = (bf16*)(ws + (64u << 20));         // 6 * 1.125 MB bf16

    for (int i = 0; i < 6; ++i)
        wcvt_kernel<<<WELEM / 256, 256, 0, stream>>>(w[i], wT + (size_t)i * WELEM);

    nchw2nhwc_kernel<<<(4 * CCH * HWS) / 256, 256, 0, stream>>>(x, xh);

    conv3x3_wmma_kernel<<<dim3(256, 2, 3), 256, 0, stream>>>(
        xh, wT, wT + WELEM, wT + 2 * (size_t)WELEM,
        sc[0], sc[1], sc[2], bi[0], bi[1], bi[2],
        Qf, Kf, Vf, 1);
    attn_kernel<<<(4 * CCH * HWS) / 256, 256, 0, stream>>>(Qf, Kf, Vf, a1h, nullptr);

    conv3x3_wmma_kernel<<<dim3(256, 2, 3), 256, 0, stream>>>(
        a1h, wT + 3 * (size_t)WELEM, wT + 4 * (size_t)WELEM, wT + 5 * (size_t)WELEM,
        sc[3], sc[4], sc[5], bi[3], bi[4], bi[5],
        Qf, Kf, Vf, 2);
    attn_kernel<<<(4 * CCH * HWS) / 256, 256, 0, stream>>>(Qf, Kf, Vf, nullptr,
                                                          (float*)d_out);
}
